// Aggregator_31456340476480
// MI455X (gfx1250) — matrix operand order repaired and checked by reading, not compiled, not run on hardware
//
#include <hip/hip_runtime.h>
#include <math.h>

typedef __attribute__((ext_vector_type(16))) __bf16 v16bf;
typedef __attribute__((ext_vector_type(8)))  float  v8f;

#define CH     64
#define HWSZ   36864           // 192*192
#define WIDTH  192
#define KTOT   576             // CH * 9, WMMA K-dim order: r = tap*64 + ci
#define XPITCH 72              // padded channel pitch in lds (halves)

struct Bf16x16 { uint4 lo, hi; };

__device__ __forceinline__ v16bf load_frag(const __bf16* p, int secondHalves) {
  Bf16x16 t;
  t.lo = *(const uint4*)(p);
  t.hi = *(const uint4*)(p + secondHalves);
  return __builtin_bit_cast(v16bf, t);
}

// CDNA5 async global->LDS copy, 16 bytes per lane, tracked by ASYNCcnt.
// VDST = VGPR holding the wave-relative LDS byte address (generic shared
// pointers carry the LDS offset in their low 32 bits).
__device__ __forceinline__ void async_copy16(const __bf16* gsrc, __bf16* ldst) {
  unsigned int l = (unsigned int)(uintptr_t)ldst;
  asm volatile("global_load_async_to_lds_b128 %0, %1, off"
               :: "v"(l), "v"(gsrc) : "memory");
}
#define WAIT_ASYNC(n) asm volatile("s_wait_asynccnt %0" :: "i"(n) : "memory")

// ---------------------------------------------------------------- pooling ---
__global__ __launch_bounds__(256) void pool_kernel(const float* __restrict__ x,
                                                   float* __restrict__ pooled) {
  const int bc = blockIdx.x;                 // 0..1023  (b*64 + c)
  const float4* p = (const float4*)(x + (size_t)bc * HWSZ);
  float s = 0.f;
  for (int i = threadIdx.x; i < HWSZ / 4; i += 256) {
    float4 v = p[i];
    s += v.x + v.y + v.z + v.w;
  }
  __shared__ float red[256];
  red[threadIdx.x] = s;
  __syncthreads();
  for (int off = 128; off > 0; off >>= 1) {
    if (threadIdx.x < off) red[threadIdx.x] += red[threadIdx.x + off];
    __syncthreads();
  }
  if (threadIdx.x == 0) pooled[bc] = red[0] * (1.f / (float)HWSZ);
}

// ------------------------------------------------- attention MLP + bias mix --
__global__ __launch_bounds__(256) void attn_kernel(
    const float* __restrict__ pooled, const float* __restrict__ w1,
    const float* __restrict__ b1, const float* __restrict__ w2,
    const float* __restrict__ b2, const float* __restrict__ Wb,
    float* __restrict__ attn_out, float* __restrict__ bdyn_out) {
  __shared__ float hsh[16 * 8];
  __shared__ float ash[16 * 4];
  const int t = threadIdx.x;
  if (t < 128) {                              // h = gelu(pooled @ w1.T + b1)
    int b = t >> 3, j = t & 7;
    float s = b1[j];
    for (int c = 0; c < CH; ++c) s += pooled[b * CH + c] * w1[j * CH + c];
    hsh[t] = 0.5f * s * (1.f + erff(s * 0.70710678118654752f));
  }
  __syncthreads();
  if (t < 64) {                               // attn = sigmoid(h @ w2.T + b2)
    int b = t >> 2, n = t & 3;
    float s = b2[n];
    for (int j = 0; j < 8; ++j) s += hsh[b * 8 + j] * w2[n * 8 + j];
    float a = 1.f / (1.f + expf(-s));
    ash[t] = a;
    attn_out[t] = a;
  }
  __syncthreads();
  for (int i = t; i < 16 * CH; i += 256) {    // bdyn = attn @ Wb
    int b = i >> 6, co = i & 63;
    float s = 0.f;
    #pragma unroll
    for (int n = 0; n < 4; ++n) s += ash[b * 4 + n] * Wb[n * CH + co];
    bdyn_out[i] = s;
  }
}

// ------------------------------------- expert-kernel mix, transposed to bf16 -
// wT[b][co][r] with r = (kh*3+kw)*64 + ci   (WMMA reduction order)
__global__ __launch_bounds__(256) void mix_kernel(const float* __restrict__ attn,
                                                  const float* __restrict__ Wk,
                                                  __bf16* __restrict__ wT) {
  int gid = blockIdx.x * 256 + threadIdx.x;
  if (gid >= 16 * CH * KTOT) return;
  int b   = gid / (CH * KTOT);
  int rem = gid % (CH * KTOT);
  int co  = rem / KTOT;
  int r   = rem % KTOT;
  int tap = r >> 6, ci = r & 63;
  const float* a = attn + b * 4;
  float s = 0.f;
  #pragma unroll
  for (int n = 0; n < 4; ++n)
    s += a[n] * Wk[(((size_t)n * CH + co) * CH + ci) * 9 + tap];
  wT[gid] = (__bf16)s;
}

// ------------------------------------------------ implicit-GEMM 3x3 conv ----
// grid = (3, 192, 16): 64-pixel tile of one output row of one sample.
// M=64 (C_out), N=64 (pixels), K=576, bf16 WMMA with f32 accumulate.
// Weights streamed tap-by-tap through a 3-deep LDS ring via async global->LDS
// copies (ASYNCcnt), fetching tap t+2 while tap t computes.
__global__ __launch_bounds__(256) void conv_kernel(
    const float* __restrict__ x, const __bf16* __restrict__ wT,
    const float* __restrict__ bdyn, float* __restrict__ out) {
  __shared__ __bf16 ldsX[3 * 66 * XPITCH];    // halo: 3 rows x 66 cols x 64ch
  __shared__ __bf16 ldsW[3][CH * XPITCH];     // async ring of per-tap weights
  __shared__ float  ldsB[CH];

  const int tid = threadIdx.x;
  const int x0  = blockIdx.x * 64;
  const int y0  = blockIdx.y;
  const int b   = blockIdx.z;
  const float* xb  = x  + (size_t)b * CH * HWSZ;
  const __bf16* wb = wT + (size_t)b * CH * KTOT;

  if (tid < CH) ldsB[tid] = bdyn[b * CH + tid];

  const int wave = tid >> 5, lane = tid & 31;

  // ---- weight async staging: each thread owns 32B of each tap slab --------
  const int wco = tid >> 2;              // 0..63
  const int wci = (tid & 3) * 16;        // 0..48
  // kick off taps 0 and 1 into ring slots 0 and 1
  #pragma unroll
  for (int t = 0; t < 2; ++t) {
    const __bf16* gp = wb + wco * KTOT + t * 64 + wci;
    __bf16* lp = &ldsW[t][wco * XPITCH + wci];
    async_copy16(gp, lp);
    async_copy16(gp + 8, lp + 8);
  }

  // ---- stage input halo (fp32 -> bf16), channel-innermost, coalesced cols --
  for (int pair = wave; pair < 192; pair += 8) {   // 64 ch x 3 rows
    int ci = pair & 63, row = pair >> 6;
    int gy = y0 + row - 1;
    bool rowok = (unsigned)gy < (unsigned)WIDTH;
    const float* src = xb + (size_t)ci * HWSZ + gy * WIDTH;
    __bf16* dst = &ldsX[(row * 66) * XPITCH + ci];
    for (int c = lane; c < 66; c += 32) {
      int gx = x0 - 1 + c;
      float v = (rowok && (unsigned)gx < (unsigned)WIDTH) ? src[gx] : 0.f;
      dst[c * XPITCH] = (__bf16)v;
    }
  }

  WAIT_ASYNC(0);                         // taps 0,1 resident
  __syncthreads();

  const int g  = lane >> 4;              // half-wave
  const int ln = lane & 15;
  const int m0 = (wave & 3) * 16;        // C_out tile
  const int p0 = (wave >> 2) * 32;       // pixel tile base (two 16-wide tiles)

  v8f acc0 = {}; v8f acc1 = {};

  #pragma unroll
  for (int t = 0; t < 9; ++t) {
    const int kh = t / 3, kw = t % 3;
    if (t + 2 <= 8) {                     // fetch tap t+2 into ring slot
      const __bf16* gp = wb + wco * KTOT + (t + 2) * 64 + wci;
      __bf16* lp = &ldsW[(t + 2) % 3][wco * XPITCH + wci];
      async_copy16(gp, lp);
      async_copy16(gp + 8, lp + 8);
    }
    const __bf16* wbuf = ldsW[t % 3];
    #pragma unroll
    for (int h = 0; h < 2; ++h) {         // two K=32 blocks per tap
      const int cb = h * 32;
      // A fragment (half-wave g holds contiguous K g*16..g*16+15)
      v16bf a = load_frag(&wbuf[(m0 + ln) * XPITCH + cb + g * 16], 8);
      // B fragments (half-wave g holds contiguous K g*16..g*16+15)
      const int col = p0 + ln + kw;
      const __bf16* xp = &ldsX[(kh * 66 + col) * XPITCH + cb + g * 16];
      v16bf bm0 = load_frag(xp, 8);
      v16bf bm1 = load_frag(xp + 16 * XPITCH, 8);
      acc0 = __builtin_amdgcn_wmma_f32_16x16x32_bf16(false, a, false, bm0,
                                                     (short)0, acc0, false, false);
      acc1 = __builtin_amdgcn_wmma_f32_16x16x32_bf16(false, a, false, bm1,
                                                     (short)0, acc1, false, false);
    }
    // retire down to "one tap (2 async ops) outstanding": proves tap t+1 done
    if (t <= 6)      WAIT_ASYNC(2);
    else if (t == 7) WAIT_ASYNC(0);
    if (t < 8) __syncthreads();           // ring slot t%3 free to overwrite
  }

  // ---- epilogue: C/D layout -> VGPR r: M = m0 + r + g*8, N = ln ------------
  const size_t obase = (size_t)b * CH * HWSZ + (size_t)y0 * WIDTH + x0 + p0 + ln;
  #pragma unroll
  for (int r = 0; r < 8; ++r) {
    int co = m0 + g * 8 + r;
    float bias = ldsB[co];
    out[obase + (size_t)co * HWSZ]      = acc0[r] + bias;
    out[obase + (size_t)co * HWSZ + 16] = acc1[r] + bias;
  }
}

// ---------------------------------------------------------------------------
extern "C" void kernel_launch(void* const* d_in, const int* in_sizes, int n_in,
                              void* d_out, int out_size, void* d_ws, size_t ws_size,
                              hipStream_t stream) {
  const float* x  = (const float*)d_in[0];
  const float* w1 = (const float*)d_in[1];
  const float* b1 = (const float*)d_in[2];
  const float* w2 = (const float*)d_in[3];
  const float* b2 = (const float*)d_in[4];
  const float* Wk = (const float*)d_in[5];
  const float* Wb = (const float*)d_in[6];
  float* out = (float*)d_out;

  char* ws = (char*)d_ws;
  float*  pooled = (float*)(ws);            // 1024 f32
  float*  attn   = (float*)(ws + 4096);     // 64 f32
  float*  bdyn   = (float*)(ws + 8192);     // 1024 f32
  __bf16* wT     = (__bf16*)(ws + 16384);   // 16*64*576 bf16 (1.15 MB)

  pool_kernel<<<16 * CH, 256, 0, stream>>>(x, pooled);
  attn_kernel<<<1, 256, 0, stream>>>(pooled, w1, b1, w2, b2, Wb, attn, bdyn);
  mix_kernel<<<(16 * CH * KTOT + 255) / 256, 256, 0, stream>>>(attn, Wk, wT);
  conv_kernel<<<dim3(3, WIDTH, 16), 256, 0, stream>>>(x, wT, bdyn, out);
}
